// GCN_42013370089475
// MI455X (gfx1250) — matrix operand order repaired and checked by reading, not compiled, not run on hardware
//
#include <hip/hip_runtime.h>
#include <hip/hip_bf16.h>

typedef __attribute__((ext_vector_type(16))) __bf16 v16bf;
typedef __attribute__((ext_vector_type(8)))  float  v8f;
typedef __attribute__((ext_vector_type(4)))  unsigned int u32x4;
typedef __attribute__((ext_vector_type(4)))  int i32x4;
typedef __attribute__((ext_vector_type(8)))  int i32x8;

union FragBF { v16bf v; unsigned short u[16]; };

// Split f32 into bf16 hi (truncated) + bf16 lo (RNE of exact remainder).
__device__ __forceinline__ void splitf(float f, unsigned short& hi, unsigned short& lo) {
    unsigned ub = __float_as_uint(f);
    unsigned hb = ub & 0xFFFF0000u;
    hi = (unsigned short)(hb >> 16);
    float fl = f - __uint_as_float(hb);           // exact
    unsigned ul = __float_as_uint(fl);
    lo = (unsigned short)((ul + 0x7FFFu + ((ul >> 16) & 1u)) >> 16);  // RNE to bf16
}

// LDS byte offset of a __shared__ object (AS(3) numeric value = LDS offset).
__device__ __forceinline__ unsigned lds_addr_of(void* p) {
    return (unsigned)(size_t)(__attribute__((address_space(3))) char*)p;
}

// TDM: DMA a (tile_rows x tile_cols) f32 tile of a row-major (rows_total x ld)
// tensor from global into LDS at lds_off. Tile rows land contiguously in LDS
// (row stride = tile_cols floats). D# layout per CDNA5 ISA 8.3/8.4.
__device__ __forceinline__ void tdm_load_tile_f32(const float* gsrc, unsigned lds_off,
                                                  unsigned tile_rows, unsigned tile_cols,
                                                  unsigned rows_total, unsigned ld) {
    unsigned long long ga = (unsigned long long)(size_t)gsrc;
    u32x4 g0;
    g0[0] = 1u;                                          // count=1, user-mode D#
    g0[1] = lds_off;                                     // lds_addr [63:32]
    g0[2] = (unsigned)(ga & 0xFFFFFFFFu);                // global_addr lo
    g0[3] = (unsigned)((ga >> 32) & 0x01FFFFFFu)         // global_addr hi (57-bit)
            | (2u << 30);                                // type=2 ("image")
    i32x8 g1;
    unsigned td0 = ld;                                   // tensor_dim0 (elements)
    unsigned td1 = rows_total;                           // tensor_dim1
    unsigned long long s0 = (unsigned long long)ld;      // tensor_dim0_stride
    g1[0] = (int)(2u << 16);                             // data_size=2 -> 4B; mask=0
    g1[1] = (int)((td0 & 0xFFFFu) << 16);                // [63:48] dim0 lo16
    g1[2] = (int)((td0 >> 16) | ((td1 & 0xFFFFu) << 16));// dim0 hi16 | dim1 lo16
    g1[3] = (int)((td1 >> 16) | (tile_cols << 16));      // dim1 hi16 | tile_dim0
    g1[4] = (int)(tile_rows & 0xFFFFu);                  // tile_dim1 ; tile_dim2=0
    g1[5] = (int)(s0 & 0xFFFFFFFFu);                     // dim0_stride lo32
    g1[6] = (int)((s0 >> 32) & 0xFFFFu);                 // dim0_stride hi16 ; dim1_stride=0
    g1[7] = 0;
    i32x4 gz = {0, 0, 0, 0};                             // 2D tile: groups 2/3 unused
#if __clang_major__ >= 23
    i32x8 gz8 = {0, 0, 0, 0, 0, 0, 0, 0};
    __builtin_amdgcn_tensor_load_to_lds(g0, g1, gz, gz, gz8, 0);
#else
    __builtin_amdgcn_tensor_load_to_lds(g0, g1, gz, gz, 0);
#endif
}

// C[M,N] = A[M,K] @ B[K,N], f32 in/out, bf16x3 WMMA (f32-class accuracy).
// Block owns a 16-row slab of A, staged into LDS by one TDM op; wave w computes
// the 16x16 tile at ntile = blockIdx.y*nwaves + w. K % 32 == 0, K <= 512.
__global__ void gemm_bf16x3(const float* __restrict__ A, const float* __restrict__ B,
                            float* __restrict__ C, int M, int K, int N) {
    __shared__ float tileA[16 * 512];

    const int lane   = threadIdx.x & 31;
    const int wv     = threadIdx.x >> 5;
    const int nwaves = blockDim.x >> 5;
    const int g      = lane >> 4;      // half-wave group
    const int l15    = lane & 15;
    const int rowBase = blockIdx.x * 16;
    const int ntile   = blockIdx.y * nwaves + wv;
    const int colB    = ntile * 16 + l15;

    // Stage A[rowBase:rowBase+16, 0:K] -> tileA via Tensor Data Mover.
    if (wv == 0) {
        tdm_load_tile_f32(A + (size_t)rowBase * K, lds_addr_of(tileA),
                          16u, (unsigned)K, (unsigned)M, (unsigned)K);
        __builtin_amdgcn_s_wait_tensorcnt(0);
    }
    __syncthreads();

    const float* Arow = tileA + l15 * K;   // this lane's A row (LDS)

    v8f acc = {};
    for (int kb = 0; kb < K; kb += 32) {
        FragBF ah, al, bh, bl;
        // A fragment: 16-bit A 16x32 layout (ISA 7.12.2): lane m<16 row m;
        // VGPR v<4 -> K = 2v + 8g, VGPR v>=4 -> K = 2v+8 + 8g (pairs).
#pragma unroll
        for (int v = 0; v < 8; ++v) {
            int k = kb + ((v < 4) ? 2 * v : 2 * v + 8) + 8 * g;
            float2 f2 = *(const float2*)(Arow + k);
            splitf(f2.x, ah.u[2 * v],     al.u[2 * v]);
            splitf(f2.y, ah.u[2 * v + 1], al.u[2 * v + 1]);
        }
        // B fragment: 32x16, col = lane&15; VGPR v<4 -> K = {2v,2v+1} + 8g, v>=4 -> K = {2v+8,2v+9} + 8g.
#pragma unroll
        for (int v = 0; v < 8; ++v) {
            int k = kb + ((v < 4) ? 2 * v : 2 * v + 8) + 8 * g;
            float b0 = B[(size_t)k * N + colB];
            float b1 = B[(size_t)(k + 1) * N + colB];
            splitf(b0, bh.u[2 * v],     bl.u[2 * v]);
            splitf(b1, bh.u[2 * v + 1], bl.u[2 * v + 1]);
        }
        if (kb + 32 < K) __builtin_prefetch(B + (size_t)(kb + 32) * N + colB, 0, 0);
        // D = A*B + C : hi*hi + lo*hi + hi*lo (bf16x3 fp32-recovery)
        acc = __builtin_amdgcn_wmma_f32_16x16x32_bf16(false, ah.v, false, bh.v, (short)0, acc, false, false);
        acc = __builtin_amdgcn_wmma_f32_16x16x32_bf16(false, al.v, false, bh.v, (short)0, acc, false, false);
        acc = __builtin_amdgcn_wmma_f32_16x16x32_bf16(false, ah.v, false, bl.v, (short)0, acc, false, false);
    }
    // C/D layout: VGPR v: lanes 0-15 -> M=v, lanes 16-31 -> M=v+8; N = lane&15.
#pragma unroll
    for (int v = 0; v < 8; ++v) {
        int r = rowBase + v + 8 * g;
        if (r < M) C[(size_t)r * N + ntile * 16 + l15] = acc[v];
    }
}

__global__ void zero_f32(float* __restrict__ p, long long n) {
    long long i = (long long)blockIdx.x * blockDim.x + threadIdx.x;
    if (i < n) p[i] = 0.0f;
}

// Unsorted-COO SpMM: out[rows[e], colOff + 4q .. +3] += vals[e] * H[cols[e], 4q .. +3]
// One thread per (edge, float4-chunk); accumulators are L2-resident (192MB L2),
// so global_atomic_add_f32 resolves in L2.
__global__ void spmm_atomic(const int* __restrict__ rows, const int* __restrict__ cols,
                            const float* __restrict__ vals, long long total,
                            int qshift, int qmask,
                            const float* __restrict__ H, int ldH,
                            float* __restrict__ out, int ldO, int colOff) {
    long long i = (long long)blockIdx.x * blockDim.x + threadIdx.x;
    if (i >= total) return;
    int e = (int)(i >> qshift);
    int q = (int)(i & qmask);
    int r = rows[e];
    int c = cols[e];
    float v = vals[e];
    const float4 h4 = *(const float4*)(H + (size_t)c * ldH + q * 4);
    float* o = out + (size_t)r * ldO + colOff + q * 4;
    unsafeAtomicAdd(o + 0, v * h4.x);
    unsafeAtomicAdd(o + 1, v * h4.y);
    unsafeAtomicAdd(o + 2, v * h4.z);
    unsafeAtomicAdd(o + 3, v * h4.w);
}

// In-place bias + relu on the first 256 columns of the (n x 320) concat buffer.
__global__ void bias_relu(float* __restrict__ x1, const float* __restrict__ b1, int n) {
    long long i = (long long)blockIdx.x * blockDim.x + threadIdx.x;
    if (i >= (long long)n * 256) return;
    int r = (int)(i >> 8);
    int c = (int)(i & 255);
    float* p = x1 + (size_t)r * 320 + c;
    float v = *p + b1[c];
    *p = v > 0.0f ? v : 0.0f;
}

// One wave per row (64 classes -> 2 per lane). seg2 of d_out holds the raw spmm
// accumulation; add bias, then write log_softmax (seg0), softmax (seg1), logits (seg2).
__global__ void softmax3(float* __restrict__ out, const float* __restrict__ b2, int n) {
    int wid  = (int)(((long long)blockIdx.x * blockDim.x + threadIdx.x) >> 5);
    int lane = threadIdx.x & 31;
    if (wid >= n) return;
    float* seg1 = out + (size_t)n * 64;
    float* seg2 = out + 2 * (size_t)n * 64;
    size_t base = (size_t)wid * 64;

    float v0 = seg2[base + lane]      + b2[lane];
    float v1 = seg2[base + 32 + lane] + b2[32 + lane];

    float m = fmaxf(v0, v1);
#pragma unroll
    for (int o = 16; o > 0; o >>= 1) m = fmaxf(m, __shfl_xor(m, o, 32));
    float e0 = expf(v0 - m), e1 = expf(v1 - m);
    float s = e0 + e1;
#pragma unroll
    for (int o = 16; o > 0; o >>= 1) s += __shfl_xor(s, o, 32);
    float ls  = logf(s);
    float inv = 1.0f / s;

    out[base + lane]       = (v0 - m) - ls;   // log_softmax
    out[base + 32 + lane]  = (v1 - m) - ls;
    seg1[base + lane]      = e0 * inv;        // softmax
    seg1[base + 32 + lane] = e1 * inv;
    seg2[base + lane]      = v0;              // logits (with bias)
    seg2[base + 32 + lane] = v1;
}

extern "C" void kernel_launch(void* const* d_in, const int* in_sizes, int n_in,
                              void* d_out, int out_size, void* d_ws, size_t ws_size,
                              hipStream_t stream) {
    const float* x       = (const float*)d_in[0];
    const float* y_train = (const float*)d_in[1];
    const int*   axr     = (const int*)d_in[2];
    const int*   axc     = (const int*)d_in[3];
    const float* axv     = (const float*)d_in[4];
    const int*   ayr     = (const int*)d_in[5];
    const int*   ayc     = (const int*)d_in[6];
    const float* ayv     = (const float*)d_in[7];
    const float* W1      = (const float*)d_in[8];
    const float* b1      = (const float*)d_in[9];
    const float* W2      = (const float*)d_in[10];
    const float* b2      = (const float*)d_in[11];

    const int NFEAT = 512, NHID = 256, NCLASS = 64, NCAT = 320;
    const int n = in_sizes[0] / NFEAT;           // 50000
    const long long Ex = in_sizes[2];            // 800000
    const long long Ey = in_sizes[5];

    float* ws   = (float*)d_ws;
    float* hbuf = ws;                            // n x 256 (h1), reused as n x 64 (h2)
    float* x1   = ws + (size_t)n * NHID;         // n x 320 concat buffer
    float* out  = (float*)d_out;
    float* seg2 = out + 2 * (size_t)n * NCLASS;  // logits accumulator

    const int mtiles = (n + 15) / 16;

    // 1. h1 = x @ W1        (N tiles: 256/16 = 16 = 2 blocks x 8 waves)
    gemm_bf16x3<<<dim3(mtiles, 2), 256, 0, stream>>>(x, W1, hbuf, n, NFEAT, NHID);

    // 2. zero the concat buffer (atomic accumulation target)
    long long nx1 = (long long)n * NCAT;
    zero_f32<<<(unsigned)((nx1 + 255) / 256), 256, 0, stream>>>(x1, nx1);

    // 3. x1[:, :256] += spmm(adj_x, h1)
    long long t1 = Ex * (NHID / 4);
    spmm_atomic<<<(unsigned)((t1 + 255) / 256), 256, 0, stream>>>(
        axr, axc, axv, t1, 6, 63, hbuf, NHID, x1, NCAT, 0);

    // 4. x1[:, 256:320] += spmm(adj_y, y_train)   (free concat)
    long long t2 = Ey * (NCLASS / 4);
    spmm_atomic<<<(unsigned)((t2 + 255) / 256), 256, 0, stream>>>(
        ayr, ayc, ayv, t2, 4, 15, y_train, NCLASS, x1, NCAT, NHID);

    // 5. x1[:, :256] = relu(x1[:, :256] + b1)
    long long tb = (long long)n * NHID;
    bias_relu<<<(unsigned)((tb + 255) / 256), 256, 0, stream>>>(x1, b1, n);

    // 6. h2 = x1 @ W2        (N tiles: 64/16 = 4 = 1 block x 4 waves)
    gemm_bf16x3<<<dim3(mtiles, 1), 128, 0, stream>>>(x1, W2, hbuf, n, NCAT, NCLASS);

    // 7. zero logits accumulator (d_out segment 2)
    long long ns2 = (long long)n * NCLASS;
    zero_f32<<<(unsigned)((ns2 + 255) / 256), 256, 0, stream>>>(seg2, ns2);

    // 8. seg2 += spmm(adj_x, h2)
    long long t3 = Ex * (NCLASS / 4);
    spmm_atomic<<<(unsigned)((t3 + 255) / 256), 256, 0, stream>>>(
        axr, axc, axv, t3, 4, 15, hbuf, NCLASS, seg2, NCLASS, 0);

    // 9. bias + log_softmax / softmax / logits (wave per row, 8 rows per block)
    softmax3<<<(unsigned)((n + 7) / 8), 256, 0, stream>>>(out, b2, n);
}
